// sLSTMCell_24154896072846
// MI455X (gfx1250) — compile-verified, matrix operand order checked by reading, not run on hardware
//
#include <hip/hip_runtime.h>
#include <cstdint>
#include <cstddef>

// ---------------------------------------------------------------------------
// sLSTM fused kernel for gfx1250 (MI455X): per-head gate GEMM via
// v_wmma_f32_16x16x32_bf16 out of LDS-staged bf16 operands, fused with the
// elementwise time recurrence held in registers (one lane per state channel).
// ---------------------------------------------------------------------------

typedef __bf16 bf16;
typedef bf16  v8bf  __attribute__((ext_vector_type(8)));
typedef bf16  v16bf __attribute__((ext_vector_type(16)));
typedef float v8f   __attribute__((ext_vector_type(8)));

#define NHEAD 4
#define HD    256
#define DMODEL 1024
#define BATCH 4
#define SEQ   2048
#define OT    32              // output-channel tile per block
#define STEP  32              // timesteps per chunk
#define KP    (HD + 8)        // padded LDS row stride in bf16 elems (264)
#define GSTR  (OT * 4 + 4)    // gate-buffer row stride in f32 (132)
#define EPSV  1e-8f

// LDS carve-up (bytes). All section bases are 16B aligned.
#define W_ELEMS   (4 * OT * KP)          // 33792 bf16
#define A_ELEMS   (4 * STEP * KP)        // 33792 bf16
#define G_OFF_B   (2 * (W_ELEMS + A_ELEMS))
#define LDS_BYTES (G_OFF_B + STEP * GSTR * 4)   // 152064 B

// Load a 16x32 bf16 fragment (A or B operand) from a padded row-major LDS
// plane, following the CDNA5 16-bit operand striping:
//   lanes 0-15  hold K = {0..7} (v0-3) and {16..23} (v4-7)
//   lanes 16-31 hold K = {8..15}       and {24..31}
// `row` = M (or N) index for this lane, `kcol` = kt*32 + half*8.
__device__ __forceinline__ v16bf frag16(const bf16* plane, int row, int kcol) {
    const bf16* p = plane + row * KP + kcol;
    v8bf lo = *reinterpret_cast<const v8bf*>(p);        // K = base .. base+7
    v8bf hi = *reinterpret_cast<const v8bf*>(p + 16);   // K = base+16 .. +23
    struct Pack { v8bf l, h; } pk{lo, hi};
    return __builtin_bit_cast(v16bf, pk);
}

__device__ __forceinline__ v8bf cvt8(const float* src) {
    float4 f0 = *reinterpret_cast<const float4*>(src);
    float4 f1 = *reinterpret_cast<const float4*>(src + 4);
    v8bf v;
    v[0] = (bf16)f0.x; v[1] = (bf16)f0.y; v[2] = (bf16)f0.z; v[3] = (bf16)f0.w;
    v[4] = (bf16)f1.x; v[5] = (bf16)f1.y; v[6] = (bf16)f1.z; v[7] = (bf16)f1.w;
    return v;
}

__global__ __launch_bounds__(128, 1)
void slstm_fused_kernel(const float* __restrict__ x,     // [B,S,4*D]
                        const float* __restrict__ W,     // [4,NH,HD,HD]
                        const float* __restrict__ bias,  // [4,NH,HD]
                        float* __restrict__ out)         // h_seq | h_f | c_f | n_f | m_f
{
    extern __shared__ char smem[];
    bf16*  sW = reinterpret_cast<bf16*>(smem);                // [4][OT][KP]
    bf16*  sA = sW + W_ELEMS;                                 // [4][STEP][KP]
    float* sG = reinterpret_cast<float*>(smem + G_OFF_B);     // [STEP][OT][4] padded

    const int otile = blockIdx.x;          // 0..HD/OT-1
    const int head  = blockIdx.y;          // 0..NHEAD-1
    const int b     = blockIdx.z;          // 0..BATCH-1
    const int tid   = threadIdx.x;
    const int wave  = tid >> 5;
    const int lane  = tid & 31;
    const int half  = lane >> 4;
    const int l16   = lane & 15;
    const int o0    = otile * OT;

    // ---- stage weights fp32 -> bf16 into LDS (one 256-wide row per thread) --
    {
        const int g = tid >> 5;            // 4 gates x 32 rows = 128 = blockDim
        const int o = tid & 31;
        const float* wr = W + (((size_t)(g * NHEAD + head) * HD + (o0 + o)) * HD);
        bf16* dst = sW + (size_t)(g * OT + o) * KP;
        #pragma unroll 4
        for (int i = 0; i < HD; i += 8)
            *reinterpret_cast<v8bf*>(dst + i) = cvt8(wr + i);
    }

    // ---- recurrent state: wave 0, lane == channel-in-tile --------------------
    float c_s = 0.f, n_s = 0.f, m_s = 0.f, h_s = 0.f;
    float bz = 0.f, bi = 0.f, bff = 0.f, bo = 0.f;
    if (tid < OT) {
        const int o = o0 + tid;
        bz  = bias[(0 * NHEAD + head) * HD + o];
        bi  = bias[(1 * NHEAD + head) * HD + o];
        bff = bias[(2 * NHEAD + head) * HD + o];
        bo  = bias[(3 * NHEAD + head) * HD + o];
    }

    __syncthreads();

    const int gfill = tid >> 5;            // gate this thread stages x for
    const int ifill = (tid & 31) * 8;      // K segment this thread stages

    for (int s0 = 0; s0 < SEQ; s0 += STEP) {
        // ---- stage x chunk fp32 -> bf16: A[g][t][i] --------------------------
        #pragma unroll 4
        for (int t = 0; t < STEP; ++t) {
            const float* xr = x + ((size_t)(b * SEQ + s0 + t) * (4 * DMODEL))
                                + gfill * DMODEL + head * HD + ifill;
            *reinterpret_cast<v8bf*>(sA + (size_t)(gfill * STEP + t) * KP + ifill) = cvt8(xr);
        }
        // token prefetch of next chunk (global_prefetch_b8)
        if (s0 + STEP < SEQ) {
            const float* pf = x + ((size_t)(b * SEQ + s0 + STEP + (tid & 31)) * (4 * DMODEL))
                                + gfill * DMODEL + head * HD;
            __builtin_prefetch(pf, 0, 1);
        }
        __syncthreads();

        // ---- GEMM: wave w owns gate w; 2x2 tiles of 16x16, K = 256 -----------
        {
            const int g = wave;
            const bf16* Ap = sA + (size_t)g * STEP * KP;
            const bf16* Wp = sW + (size_t)g * OT * KP;
            #pragma unroll
            for (int mt = 0; mt < 2; ++mt) {
                #pragma unroll
                for (int nt = 0; nt < 2; ++nt) {
                    v8f acc = {};
                    #pragma unroll
                    for (int kt = 0; kt < 8; ++kt) {
                        v16bf af = frag16(Ap, mt * 16 + l16, kt * 32 + half * 8);
                        v16bf bf = frag16(Wp, nt * 16 + l16, kt * 32 + half * 8);
                        acc = __builtin_amdgcn_wmma_f32_16x16x32_bf16(
                                  false, af, false, bf, (short)0, acc, false, false);
                    }
                    // C layout: lane<16 -> M=r, lane>=16 -> M=r+8 ; N = lane%16
                    const int t_base = mt * 16 + half * 8;
                    const int o_out  = nt * 16 + l16;
                    #pragma unroll
                    for (int r = 0; r < 8; ++r)
                        sG[(size_t)(t_base + r) * GSTR + o_out * 4 + g] = acc[r];
                }
            }
        }
        __syncthreads();

        // ---- scan: wave 0 lanes advance their channel through STEP steps -----
        if (tid < OT) {
            const int o = tid;
            float* hout = out + (((size_t)(b * SEQ + s0) * NHEAD + head) * HD) + o0 + o;
            for (int t = 0; t < STEP; ++t) {
                float4 gz = *reinterpret_cast<const float4*>(sG + (size_t)t * GSTR + o * 4);
                float zb = gz.x + bz, ib = gz.y + bi, fb = gz.z + bff, ob = gz.w + bo;
                float zv = tanhf(zb);
                float ov = 1.f / (1.f + __expf(-ob));
                float lf = __logf(__expf(fb) + EPSV);
                float li = __logf(__expf(ib) + EPSV);
                float mn = fmaxf(lf + m_s, li);
                float ih = __expf(li - mn);
                float fh = __expf(lf + m_s - mn);
                c_s = fh * c_s + ih * zv;
                n_s = fh * n_s + ih;
                m_s = mn;
                h_s = ov * (c_s / (n_s + EPSV));
                hout[(size_t)t * (NHEAD * HD)] = h_s;
            }
        }
        __syncthreads();
    }

    // ---- final states: d_out = h_seq | h_f | c_f | n_f | m_f -----------------
    if (tid < OT) {
        const size_t base = (size_t)BATCH * SEQ * NHEAD * HD;
        const size_t st   = (size_t)BATCH * NHEAD * HD;
        const size_t idx  = ((size_t)b * NHEAD + head) * HD + o0 + tid;
        out[base + 0 * st + idx] = h_s;
        out[base + 1 * st + idx] = c_s;
        out[base + 2 * st + idx] = n_s;
        out[base + 3 * st + idx] = m_s;
    }
}

extern "C" void kernel_launch(void* const* d_in, const int* in_sizes, int n_in,
                              void* d_out, int out_size, void* d_ws, size_t ws_size,
                              hipStream_t stream) {
    (void)in_sizes; (void)n_in; (void)out_size; (void)d_ws; (void)ws_size;
    const float* x    = (const float*)d_in[0];   // [4,2048,4096]
    const float* W    = (const float*)d_in[1];   // [4,4,256,256]
    const float* bias = (const float*)d_in[2];   // [4,4,256]
    float* out = (float*)d_out;

    dim3 grid(HD / OT, NHEAD, BATCH);            // 8 x 4 x 4 = 128 blocks
    slstm_fused_kernel<<<grid, 128, LDS_BYTES, stream>>>(x, W, bias, out);
}
